// SnapshotEncoder_36326833390308
// MI455X (gfx1250) — compile-verified, matrix operand order checked by reading, not run on hardware
//
#include <hip/hip_runtime.h>
#include <hip/hip_bf16.h>
#include <limits>

// ---------------------------------------------------------------------------
// Problem constants (from the reference)
// ---------------------------------------------------------------------------
static constexpr int NN  = 50000;   // nodes
static constexpr int NE  = 800000;  // edges
static constexpr int ND  = 128;     // node feature dim
static constexpr int ED  = 32;      // edge feature dim
static constexpr int HID = 64;      // per-head channels
static constexpr int HD1 = 2;       // heads, layer 1
static constexpr int F1  = HD1 * HID; // 128
static constexpr float LN_EPS = 1e-5f;

typedef __attribute__((ext_vector_type(16))) _Float16 v16h;
typedef __attribute__((ext_vector_type(8)))  _Float16 v8h;
typedef __attribute__((ext_vector_type(4)))  _Float16 v4h;
typedef __attribute__((ext_vector_type(8)))  float    v8f;

// ---------------------------------------------------------------------------
// Small utility kernels
// ---------------------------------------------------------------------------
__global__ void fill_f32_kernel(float* __restrict__ p, float v, int n) {
    int i = blockIdx.x * blockDim.x + threadIdx.x;
    if (i < n) p[i] = v;
}

__global__ void cvt_f16_kernel(const float* __restrict__ in, _Float16* __restrict__ out, int n) {
    int i = blockIdx.x * blockDim.x + threadIdx.x;
    if (i < n) out[i] = (_Float16)in[i];
}

// W[K][N] f32 (row-major)  ->  Wt[N][K] f16 (row-major) so WMMA B fragments
// load contiguous-K just like A fragments.
__global__ void wtrans_kernel(const float* __restrict__ W, _Float16* __restrict__ Wt,
                              int K, int N) {
    int i = blockIdx.x * blockDim.x + threadIdx.x;
    if (i >= K * N) return;
    int k = i / N, n = i - k * N;
    Wt[n * K + k] = (_Float16)W[k * N + n];
}

// ---------------------------------------------------------------------------
// WMMA GEMM (wave32, one 16x16 C tile per wave, K stepped by 32)
// A: [M][K] f16 row-major.  Bt: [N][K] f16 row-major (i.e. B transposed).
//
// 16-bit A-fragment layout (ISA 7.12.2): lane l -> m = l&15, kh = l>>4;
//   elements 0..7  = K (kh*8 .. kh*8+7)
//   elements 8..15 = K (16+kh*8 .. 16+kh*8+7)
// B uses the mirrored layout with n = l&15, so loading from Bt is identical.
// ---------------------------------------------------------------------------
__device__ __forceinline__ v16h load_frag(const _Float16* __restrict__ base, int ld,
                                          int row0, int k0, int lane) {
    int m  = lane & 15;
    int kh = lane >> 4;
    const _Float16* p = base + (size_t)(row0 + m) * ld + k0 + kh * 8;
    v8h lo = *(const v8h*)p;
    v8h hi = *(const v8h*)(p + 16);
    v16h r;
#pragma unroll
    for (int i = 0; i < 8; ++i) { r[i] = lo[i]; r[i + 8] = hi[i]; }
    return r;
}

// C1 = A @ B1, C2 = A @ B2 (shares A fragment between the two WMMAs).
__global__ void gemm_dualB_kernel(const _Float16* __restrict__ A,
                                  const _Float16* __restrict__ B1t,
                                  const _Float16* __restrict__ B2t,
                                  float* __restrict__ C1, float* __restrict__ C2,
                                  int M, int N, int K) {
    int lane = threadIdx.x & 31;
    int wid  = (blockIdx.x * blockDim.x + threadIdx.x) >> 5;
    int tn = N >> 4;
    int tiles = (M >> 4) * tn;
    if (wid >= tiles) return;                 // uniform per wave
    int tm = wid / tn, tc = wid - tm * tn;
    int row0 = tm << 4, col0 = tc << 4;
    v8f acc1 = {}; v8f acc2 = {};
    for (int k0 = 0; k0 < K; k0 += 32) {
        v16h a  = load_frag(A,   K, row0, k0, lane);
        v16h b1 = load_frag(B1t, K, col0, k0, lane);
        acc1 = __builtin_amdgcn_wmma_f32_16x16x32_f16(false, a, false, b1,
                                                      (short)0, acc1, false, false);
        v16h b2 = load_frag(B2t, K, col0, k0, lane);
        acc2 = __builtin_amdgcn_wmma_f32_16x16x32_f16(false, a, false, b2,
                                                      (short)0, acc2, false, false);
    }
    int n = lane & 15, kh = lane >> 4;
#pragma unroll
    for (int v = 0; v < 8; ++v) {             // lanes 0..15 store 64B contiguous
        size_t idx = (size_t)(row0 + v + 8 * kh) * N + col0 + n;
        C1[idx] = acc1[v];
        C2[idx] = acc2[v];
    }
}

// C = A @ B with f16 output (used for the big streamed edge tensor xe).
__global__ void gemm_f16out_kernel(const _Float16* __restrict__ A,
                                   const _Float16* __restrict__ Bt,
                                   _Float16* __restrict__ C,
                                   int M, int N, int K) {
    int lane = threadIdx.x & 31;
    int wid  = (blockIdx.x * blockDim.x + threadIdx.x) >> 5;
    int tn = N >> 4;
    int tiles = (M >> 4) * tn;
    if (wid >= tiles) return;
    int tm = wid / tn, tc = wid - tm * tn;
    int row0 = tm << 4, col0 = tc << 4;
    v8f acc = {};
    for (int k0 = 0; k0 < K; k0 += 32) {
        v16h a = load_frag(A,  K, row0, k0, lane);
        v16h b = load_frag(Bt, K, col0, k0, lane);
        acc = __builtin_amdgcn_wmma_f32_16x16x32_f16(false, a, false, b,
                                                     (short)0, acc, false, false);
    }
    int n = lane & 15, kh = lane >> 4;
#pragma unroll
    for (int v = 0; v < 8; ++v) {
        C[(size_t)(row0 + v + 8 * kh) * N + col0 + n] = (_Float16)acc[v];
    }
}

// ---------------------------------------------------------------------------
// Edge pass 1: logits[e][h] = att_h . leaky_relu(xl[src] + xr[dst] + xe[e]);
// segment-max over dst via signed/unsigned int atomic trick (segmax init -inf).
// ---------------------------------------------------------------------------
__device__ __forceinline__ void atomicMaxFloat(float* addr, float val) {
    if (val >= 0.f) atomicMax((int*)addr, __float_as_int(val));
    else            atomicMin((unsigned int*)addr, __float_as_uint(val));
}

template <int H, int C>
__global__ void edge_logits_kernel(const float* __restrict__ xl,
                                   const float* __restrict__ xr,
                                   const _Float16* __restrict__ xe,
                                   const int* __restrict__ srcI,
                                   const int* __restrict__ dstI,
                                   const float* __restrict__ att,
                                   float* __restrict__ logits,
                                   float* __restrict__ segmax, int E) {
    constexpr int F = H * C;
    __shared__ float s_att[F];
    for (int i = threadIdx.x; i < F; i += blockDim.x) s_att[i] = att[i];
    __syncthreads();
    int e = blockIdx.x * blockDim.x + threadIdx.x;
    if (e >= E) return;
    int s = srcI[e], d = dstI[e];
    const float4* pl = (const float4*)(xl + (size_t)s * F);   // L2-resident gather
    const float4* pr = (const float4*)(xr + (size_t)d * F);
    const v4h*    pe = (const v4h*)(xe + (size_t)e * F);      // streamed f16
    float acc[H];
#pragma unroll
    for (int h = 0; h < H; ++h) acc[h] = 0.f;
#pragma unroll
    for (int j = 0; j < F / 4; ++j) {
        float4 a = pl[j];
        float4 b = pr[j];
        v4h    ev = pe[j];
        int h = (j * 4) / C;  // compile-time per unrolled iter
        float m0 = a.x + b.x + (float)ev[0];
        float m1 = a.y + b.y + (float)ev[1];
        float m2 = a.z + b.z + (float)ev[2];
        float m3 = a.w + b.w + (float)ev[3];
        m0 = m0 > 0.f ? m0 : 0.2f * m0;
        m1 = m1 > 0.f ? m1 : 0.2f * m1;
        m2 = m2 > 0.f ? m2 : 0.2f * m2;
        m3 = m3 > 0.f ? m3 : 0.2f * m3;
        acc[h] += s_att[j * 4 + 0] * m0 + s_att[j * 4 + 1] * m1 +
                  s_att[j * 4 + 2] * m2 + s_att[j * 4 + 3] * m3;
    }
#pragma unroll
    for (int h = 0; h < H; ++h) {
        logits[(size_t)e * H + h] = acc[h];
        atomicMaxFloat(&segmax[(size_t)d * H + h], acc[h]);
    }
}

// ---------------------------------------------------------------------------
// Edge pass 2: w = exp(logit - segmax[dst]); denom[dst] += w;
//              accum[dst] += w * xl[src]   (f32 L2 atomics)
// ---------------------------------------------------------------------------
template <int H, int C>
__global__ void edge_accum_kernel(const float* __restrict__ xl,
                                  const int* __restrict__ srcI,
                                  const int* __restrict__ dstI,
                                  const float* __restrict__ logits,
                                  const float* __restrict__ segmax,
                                  float* __restrict__ denom,
                                  float* __restrict__ accum, int E) {
    constexpr int F = H * C;
    int e = blockIdx.x * blockDim.x + threadIdx.x;
    if (e >= E) return;
    int s = srcI[e], d = dstI[e];
    float w[H];
#pragma unroll
    for (int h = 0; h < H; ++h) {
        w[h] = expf(logits[(size_t)e * H + h] - segmax[(size_t)d * H + h]);
        atomicAdd(&denom[(size_t)d * H + h], w[h]);
    }
    const float4* pl = (const float4*)(xl + (size_t)s * F);
    float* pa = accum + (size_t)d * F;
#pragma unroll
    for (int j = 0; j < F / 4; ++j) {
        float4 a = pl[j];
        int h = (j * 4) / C;
        atomicAdd(&pa[j * 4 + 0], w[h] * a.x);
        atomicAdd(&pa[j * 4 + 1], w[h] * a.y);
        atomicAdd(&pa[j * 4 + 2], w[h] * a.z);
        atomicAdd(&pa[j * 4 + 3], w[h] * a.w);
    }
}

// ---------------------------------------------------------------------------
// Node finalize: out = accum/(denom+1e-16) + bias -> LayerNorm -> ReLU.
// One wave32 per node; LN reduction via __shfl_xor. Optionally emits the
// f16 copy that feeds the next layer's WMMA GEMM.
// ---------------------------------------------------------------------------
template <int F, int H>
__global__ void node_finalize_kernel(const float* __restrict__ accum,
                                     const float* __restrict__ denom,
                                     const float* __restrict__ bias,
                                     const float* __restrict__ gamma,
                                     const float* __restrict__ beta,
                                     _Float16* __restrict__ out16,
                                     float* __restrict__ out32, int N) {
    constexpr int C  = F / H;
    constexpr int PL = F / 32;   // features per lane
    int lane = threadIdx.x & 31;
    int node = (blockIdx.x * blockDim.x + threadIdx.x) >> 5;
    if (node >= N) return;
    float v[PL];
    float sum = 0.f;
#pragma unroll
    for (int j = 0; j < PL; ++j) {
        int f = lane + j * 32;               // coalesced across the wave
        int h = f / C;
        float x = accum[(size_t)node * F + f] /
                  (denom[(size_t)node * H + h] + 1e-16f) + bias[f];
        v[j] = x;
        sum += x;
    }
#pragma unroll
    for (int o = 16; o > 0; o >>= 1) sum += __shfl_xor(sum, o, 32);
    float mu = sum * (1.0f / F);
    float vs = 0.f;
#pragma unroll
    for (int j = 0; j < PL; ++j) { float t = v[j] - mu; vs += t * t; }
#pragma unroll
    for (int o = 16; o > 0; o >>= 1) vs += __shfl_xor(vs, o, 32);
    float inv = rsqrtf(vs * (1.0f / F) + LN_EPS);
#pragma unroll
    for (int j = 0; j < PL; ++j) {
        int f = lane + j * 32;
        float y = (v[j] - mu) * inv * gamma[f] + beta[f];
        y = fmaxf(y, 0.f);
        if (out32) out32[(size_t)node * F + f] = y;
        if (out16) out16[(size_t)node * F + f] = (_Float16)y;
    }
}

// ---------------------------------------------------------------------------
// Host-side launcher
// ---------------------------------------------------------------------------
extern "C" void kernel_launch(void* const* d_in, const int* in_sizes, int n_in,
                              void* d_out, int out_size, void* d_ws, size_t ws_size,
                              hipStream_t stream) {
    (void)in_sizes; (void)n_in; (void)out_size; (void)ws_size;
    const float* x    = (const float*)d_in[0];
    const int*   ei   = (const int*)d_in[1];
    const float* ea   = (const float*)d_in[2];
    const float* Wl1  = (const float*)d_in[3];
    const float* Wr1  = (const float*)d_in[4];
    const float* We1  = (const float*)d_in[5];
    const float* att1 = (const float*)d_in[6];
    const float* b1   = (const float*)d_in[7];
    const float* g1   = (const float*)d_in[8];
    const float* be1  = (const float*)d_in[9];
    const float* Wl2  = (const float*)d_in[10];
    const float* Wr2  = (const float*)d_in[11];
    const float* We2  = (const float*)d_in[12];
    const float* att2 = (const float*)d_in[13];
    const float* b2   = (const float*)d_in[14];
    const float* g2   = (const float*)d_in[15];
    const float* be2  = (const float*)d_in[16];
    const int* srcI = ei;
    const int* dstI = ei + NE;
    float* out = (float*)d_out;

    // ---- carve workspace (fixed offsets, ~353 MB total) ----
    char* wp = (char*)d_ws;
    auto carve = [&](size_t bytes) {
        void* p = wp; wp += (bytes + 255) & ~(size_t)255; return p;
    };
    _Float16* aF16 = (_Float16*)carve((size_t)NN * F1 * 2);   // x f16, later h1 f16
    _Float16* eF16 = (_Float16*)carve((size_t)NE * ED * 2);
    _Float16* Wl1t = (_Float16*)carve((size_t)F1 * ND * 2);
    _Float16* Wr1t = (_Float16*)carve((size_t)F1 * ND * 2);
    _Float16* We1t = (_Float16*)carve((size_t)F1 * ED * 2);
    _Float16* Wl2t = (_Float16*)carve((size_t)HID * F1 * 2);
    _Float16* Wr2t = (_Float16*)carve((size_t)HID * F1 * 2);
    _Float16* We2t = (_Float16*)carve((size_t)HID * ED * 2);
    float*    xl    = (float*)carve((size_t)NN * F1 * 4);
    float*    xr    = (float*)carve((size_t)NN * F1 * 4);
    _Float16* xe16  = (_Float16*)carve((size_t)NE * F1 * 2);
    float*    logits = (float*)carve((size_t)NE * HD1 * 4);
    float*    segmax = (float*)carve((size_t)NN * HD1 * 4);
    float*    denom  = (float*)carve((size_t)NN * HD1 * 4);
    float*    accum  = (float*)carve((size_t)NN * F1 * 4);

    const float NEG_INF = -std::numeric_limits<float>::infinity();
    auto blocks = [](long long n, int t) { return (int)((n + t - 1) / t); };

    // ---- prep: f16 conversions + transposed f16 weights ----
    cvt_f16_kernel<<<blocks((long long)NN * ND, 256), 256, 0, stream>>>(x, aF16, NN * ND);
    cvt_f16_kernel<<<blocks((long long)NE * ED, 256), 256, 0, stream>>>(ea, eF16, NE * ED);
    wtrans_kernel<<<blocks(ND * F1, 256), 256, 0, stream>>>(Wl1, Wl1t, ND, F1);
    wtrans_kernel<<<blocks(ND * F1, 256), 256, 0, stream>>>(Wr1, Wr1t, ND, F1);
    wtrans_kernel<<<blocks(ED * F1, 256), 256, 0, stream>>>(We1, We1t, ED, F1);
    wtrans_kernel<<<blocks(F1 * HID, 256), 256, 0, stream>>>(Wl2, Wl2t, F1, HID);
    wtrans_kernel<<<blocks(F1 * HID, 256), 256, 0, stream>>>(Wr2, Wr2t, F1, HID);
    wtrans_kernel<<<blocks(ED * HID, 256), 256, 0, stream>>>(We2, We2t, ED, HID);

    // =========================== Layer 1 (H=2, C=64) ===========================
    fill_f32_kernel<<<blocks(NN * HD1, 256), 256, 0, stream>>>(segmax, NEG_INF, NN * HD1);
    fill_f32_kernel<<<blocks(NN * HD1, 256), 256, 0, stream>>>(denom, 0.f, NN * HD1);
    fill_f32_kernel<<<blocks((long long)NN * F1, 256), 256, 0, stream>>>(accum, 0.f, NN * F1);

    {   // xl = x@Wl1, xr = x@Wr1  (M=50000, N=128, K=128)
        int tiles = (NN / 16) * (F1 / 16);
        gemm_dualB_kernel<<<blocks((long long)tiles * 32, 128), 128, 0, stream>>>(
            aF16, Wl1t, Wr1t, xl, xr, NN, F1, ND);
    }
    {   // xe = edge_attr@We1 (M=800000, N=128, K=32) -> f16
        int tiles = (NE / 16) * (F1 / 16);
        gemm_f16out_kernel<<<blocks((long long)tiles * 32, 128), 128, 0, stream>>>(
            eF16, We1t, xe16, NE, F1, ED);
    }
    edge_logits_kernel<HD1, HID><<<blocks(NE, 256), 256, 0, stream>>>(
        xl, xr, xe16, srcI, dstI, att1, logits, segmax, NE);
    edge_accum_kernel<HD1, HID><<<blocks(NE, 256), 256, 0, stream>>>(
        xl, srcI, dstI, logits, segmax, denom, accum, NE);
    // finalize layer 1 -> h1 (f16 only; it is consumed only by layer-2 GEMM)
    node_finalize_kernel<F1, HD1><<<blocks((long long)NN * 32, 128), 128, 0, stream>>>(
        accum, denom, b1, g1, be1, aF16, (float*)nullptr, NN);

    // =========================== Layer 2 (H=1, C=64) ===========================
    fill_f32_kernel<<<blocks(NN, 256), 256, 0, stream>>>(segmax, NEG_INF, NN);
    fill_f32_kernel<<<blocks(NN, 256), 256, 0, stream>>>(denom, 0.f, NN);
    fill_f32_kernel<<<blocks((long long)NN * HID, 256), 256, 0, stream>>>(accum, 0.f, NN * HID);

    {   // xl2 = h1@Wl2, xr2 = h1@Wr2  (M=50000, N=64, K=128)
        int tiles = (NN / 16) * (HID / 16);
        gemm_dualB_kernel<<<blocks((long long)tiles * 32, 128), 128, 0, stream>>>(
            aF16, Wl2t, Wr2t, xl, xr, NN, HID, F1);
    }
    {   // xe2 = edge_attr@We2 (M=800000, N=64, K=32) -> f16
        int tiles = (NE / 16) * (HID / 16);
        gemm_f16out_kernel<<<blocks((long long)tiles * 32, 128), 128, 0, stream>>>(
            eF16, We2t, xe16, NE, HID, ED);
    }
    edge_logits_kernel<1, HID><<<blocks(NE, 256), 256, 0, stream>>>(
        xl, xr, xe16, srcI, dstI, att2, logits, segmax, NE);
    edge_accum_kernel<1, HID><<<blocks(NE, 256), 256, 0, stream>>>(
        xl, srcI, dstI, logits, segmax, denom, accum, NE);
    // finalize layer 2 -> d_out (f32)
    node_finalize_kernel<HID, 1><<<blocks((long long)NN * 32, 128), 128, 0, stream>>>(
        accum, denom, b2, g2, be2, (_Float16*)nullptr, out, NN);
}
